// Detector_41635412968063
// MI455X (gfx1250) — matrix operand order repaired and checked by reading, not compiled, not run on hardware
//
#include <hip/hip_runtime.h>
#include <hip/hip_bf16.h>

typedef __attribute__((ext_vector_type(16))) _Float16 v16h;
typedef __attribute__((ext_vector_type(8)))  float    v8f;

typedef unsigned int tdm_u32x4 __attribute__((ext_vector_type(4)));
typedef int          tdm_i32x4 __attribute__((ext_vector_type(4)));
typedef int          tdm_i32x8 __attribute__((ext_vector_type(8)));

#define B_  4
#define N_  16384
#define M_  512
#define K_  32
#define NK_ 64
#define KG_ 8
#define C3_ 256

// ---------------------------------------------------------------------------
// 1) Build combined EdgeConv weights in f16:  W' = [W_top | W_bot - W_top]
//    conv1: (12->32 pad) x 128, conv2: 64 x 512
// ---------------------------------------------------------------------------
__global__ void prep_weights_kernel(const float* __restrict__ we1,
                                    const float* __restrict__ we2,
                                    _Float16* __restrict__ w1p,
                                    _Float16* __restrict__ w2p) {
    int i = blockIdx.x * blockDim.x + threadIdx.x;
    if (i < 32 * 128) {
        int k = i >> 7, n = i & 127;
        float v = 0.0f;
        if (k < 12)
            v = (n < 64) ? we1[k * 64 + n]
                         : (we1[(12 + k) * 64 + (n - 64)] - we1[k * 64 + (n - 64)]);
        w1p[i] = (_Float16)v;
    }
    int j = i - 32 * 128;
    if (j >= 0 && j < 64 * 512) {
        int k = j >> 9, n = j & 511;
        float v = (n < 256) ? we2[k * 256 + n]
                            : (we2[(64 + k) * 256 + (n - 256)] - we2[k * 256 + (n - 256)]);
        w2p[j] = (_Float16)v;
    }
}

// ---------------------------------------------------------------------------
// 2) Farthest point sampling: one block per batch, serial 512 iterations.
// ---------------------------------------------------------------------------
__global__ __launch_bounds__(1024) void fps_kernel(const float4* __restrict__ x,
                                                   int* __restrict__ node_idx) {
    __shared__ float sval[32];
    __shared__ int   sidx[32];
    __shared__ float pp[3];
    __shared__ int   scur;
    const int b   = blockIdx.x;
    const int tid = threadIdx.x;
    const int wv  = tid >> 5;
    const int ln  = tid & 31;
    const float4* xb = x + (size_t)b * N_;

    float px[16], py[16], pz[16], md[16];
#pragma unroll
    for (int j = 0; j < 16; ++j) {
        int n = (j << 10) | tid;
        float4 v = xb[n];
        px[j] = v.x; py[j] = v.y; pz[j] = v.z; md[j] = 1e10f;
    }
    if (tid == 0) scur = 0;
    __syncthreads();

    for (int it = 0; it < M_; ++it) {
        if (tid == 0) {
            int cur = scur;
            node_idx[b * M_ + it] = cur;
            float4 v = xb[cur];
            pp[0] = v.x; pp[1] = v.y; pp[2] = v.z;
        }
        __syncthreads();
        float qx = pp[0], qy = pp[1], qz = pp[2];
        float bv = -1.0f; int bi = 0;
#pragma unroll
        for (int j = 0; j < 16; ++j) {
            int n = (j << 10) | tid;
            float dx = px[j] - qx, dy = py[j] - qy, dz = pz[j] - qz;
            float d = dx * dx + dy * dy + dz * dz;
            float mm = fminf(md[j], d);
            md[j] = mm;
            if (mm > bv) { bv = mm; bi = n; }   // ascending n -> first max kept
        }
        for (int o = 16; o > 0; o >>= 1) {
            float ov = __shfl_xor(bv, o);
            int   oi = __shfl_xor(bi, o);
            if (ov > bv || (ov == bv && oi < bi)) { bv = ov; bi = oi; }
        }
        if (ln == 0) { sval[wv] = bv; sidx[wv] = bi; }
        __syncthreads();
        if (wv == 0) {
            float v2 = sval[ln]; int i2 = sidx[ln];
            for (int o = 16; o > 0; o >>= 1) {
                float ov = __shfl_xor(v2, o);
                int   oi = __shfl_xor(i2, o);
                if (ov > v2 || (ov == v2 && oi < i2)) { v2 = ov; i2 = oi; }
            }
            if (ln == 0) scur = i2;
        }
        __syncthreads();
    }
}

// ---------------------------------------------------------------------------
// 3) Gather node features
// ---------------------------------------------------------------------------
__global__ void gather_node_kernel(const float4* __restrict__ x,
                                   const int* __restrict__ node_idx,
                                   float4* __restrict__ node) {
    int i = blockIdx.x * blockDim.x + threadIdx.x;
    if (i < B_ * M_) {
        int b = i >> 9;
        node[i] = x[(size_t)b * N_ + node_idx[i]];
    }
}

// ---------------------------------------------------------------------------
// 4) Top-64 nearest points per node, then select K=32 via rand_idx.
// ---------------------------------------------------------------------------
__global__ __launch_bounds__(256) void knn64_kernel(const float4* __restrict__ x,
                                                    const float4* __restrict__ node,
                                                    const int* __restrict__ rand_idx,
                                                    int* __restrict__ sel) {
    extern __shared__ float d2[];  // N_ floats (64 KB)
    __shared__ float rv[8];
    __shared__ int   ri[8];
    __shared__ int   nn[NK_];
    const int g   = blockIdx.x;
    const int b   = g >> 9;
    const int tid = threadIdx.x;
    const int wv  = tid >> 5;
    const int ln  = tid & 31;
    float4 nd = node[g];
    float sn = nd.x * nd.x + nd.y * nd.y + nd.z * nd.z;
    const float4* xb = x + (size_t)b * N_;
    for (int i = 0; i < 64; ++i) {
        int n = (i << 8) | tid;
        float4 v = xb[n];
        float sp = v.x * v.x + v.y * v.y + v.z * v.z;
        d2[n] = sn + sp - 2.0f * (nd.x * v.x + nd.y * v.y + nd.z * v.z);
    }
    __syncthreads();
    for (int s = 0; s < NK_; ++s) {
        float bv = 3e38f; int bi = 0;
        for (int i = 0; i < 64; ++i) {
            int n = (i << 8) | tid;
            float v = d2[n];
            if (v < bv) { bv = v; bi = n; }
        }
        for (int o = 16; o > 0; o >>= 1) {
            float ov = __shfl_xor(bv, o);
            int   oi = __shfl_xor(bi, o);
            if (ov < bv || (ov == bv && oi < bi)) { bv = ov; bi = oi; }
        }
        if (ln == 0) { rv[wv] = bv; ri[wv] = bi; }
        __syncthreads();
        if (tid == 0) {
            float v = rv[0]; int idx = ri[0];
            for (int w = 1; w < 8; ++w)
                if (rv[w] < v || (rv[w] == v && ri[w] < idx)) { v = rv[w]; idx = ri[w]; }
            nn[s] = idx;
            d2[idx] = 3e38f;
        }
        __syncthreads();
    }
    if (tid < K_) sel[g * K_ + tid] = nn[rand_idx[tid]];
}

// ---------------------------------------------------------------------------
// 5) Build 12-channel cluster features + write transposed random_cluster out.
// ---------------------------------------------------------------------------
__global__ void cluster_kernel(const float4* __restrict__ x,
                               const float4* __restrict__ node,
                               const int* __restrict__ sel,
                               float* __restrict__ clus,
                               float* __restrict__ rc) {
    int i = blockIdx.x * blockDim.x + threadIdx.x;
    if (i >= B_ * M_ * K_) return;
    int g = i >> 5, k = i & 31;
    int b = g >> 9, m = g & 511;
    float4 nb = x[(size_t)b * N_ + sel[i]];
    float4 nd = node[g];
    float rx = nb.x - nd.x, ry = nb.y - nd.y, rz = nb.z - nd.z;
    float dist = sqrtf(rx * rx + ry * ry + rz * rz);
    float f[12] = {nd.x, nd.y, nd.z, nd.w, nb.x, nb.y, nb.z, nb.w, rx, ry, rz, dist};
    float* cr = clus + (size_t)i * 12;
#pragma unroll
    for (int c = 0; c < 12; ++c) {
        cr[c] = f[c];
        rc[(((size_t)b * 12 + c) * M_ + m) * K_ + k] = f[c];
    }
}

// ---------------------------------------------------------------------------
// 6) EdgeConv: Y = X_f16(32 x CPAD) @ W'(CPAD x 2D) via v_wmma_f32_16x16x32_f16,
//    then out[p,d] = max_{j in knn8(p)} relu(g*(Y[j,d] + Y[p,D+d]) + b).
//    When CIN==CPAD the (contiguous) input tile is staged into LDS with the
//    Tensor Data Mover (TENSOR_LOAD_TO_LDS + s_wait_tensorcnt).
// ---------------------------------------------------------------------------
template <int CIN, int CPAD, int DOUT, int OUTMODE>
__global__ __launch_bounds__(256) void edgeconv_kernel(const float* __restrict__ src,
                                                       const _Float16* __restrict__ Wp,
                                                       const float* __restrict__ gamma,
                                                       const float* __restrict__ beta,
                                                       float* __restrict__ dst) {
    extern __shared__ char smem[];
    const int TWO_D = 2 * DOUT;
    float*    Xf = (float*)smem;                   // 32*CPAD f32 (at dynamic-LDS base)
    _Float16* Xh = (_Float16*)(Xf + 32 * CPAD);    // 32*CPAD f16
    float*    Y  = (float*)(Xh + 32 * CPAD);       // 32*TWO_D f32
    float*    dd = Y + 32 * TWO_D;                 // 32*32 f32
    int*      kn = (int*)(dd + 32 * 32);           // 32*8 int
    const int g   = blockIdx.x;
    const int tid = threadIdx.x;

    if (tid == 0) __builtin_prefetch(Wp, 0, 1);    // global_prefetch_b8 toward L2

    if (CIN == CPAD) {
        // ---- TDM: DMA the contiguous 32*CPAD f32 tile straight into LDS ----
        if (tid < 32) {  // wave 0 issues exactly one TENSOR_LOAD_TO_LDS
            const unsigned nelem = 32u * (unsigned)CPAD;  // f32 elements (fits 16b)
            unsigned long long ga =
                (unsigned long long)(uintptr_t)(src + (size_t)g * 32 * CIN);
            unsigned lds_base = (unsigned)__builtin_amdgcn_groupstaticsize();
            tdm_u32x4 g0;
            g0[0] = 1u;                                   // count=1 (user mode)
            g0[1] = lds_base;                             // lds_addr = Xf
            g0[2] = (unsigned)(ga & 0xffffffffull);       // global_addr[31:0]
            g0[3] = (unsigned)((ga >> 32) & 0x01ffffffull) | 0x80000000u; // type=2
            tdm_i32x8 g1;
            g1[0] = (int)(2u << 16);                      // data_size = 4 bytes
            g1[1] = (int)((nelem & 0xffffu) << 16);       // tensor_dim0[15:0]
            g1[2] = (int)((nelem >> 16) | (1u << 16));    // dim0[31:16] | tensor_dim1=1
            g1[3] = (int)((nelem & 0xffffu) << 16);       // tile_dim0 = nelem
            g1[4] = 1;                                    // tile_dim1=1, tile_dim2=0
            g1[5] = (int)nelem;                           // tensor_dim0_stride lo32
            g1[6] = 0;                                    // stride hi / dim1_stride lo
            g1[7] = 0;
            tdm_i32x4 z4 = {0, 0, 0, 0};
#if defined(__clang_major__) && (__clang_major__ >= 23)
            tdm_i32x8 z8 = {0, 0, 0, 0, 0, 0, 0, 0};
            __builtin_amdgcn_tensor_load_to_lds(g0, g1, z4, z4, z8, 0);
#else
            __builtin_amdgcn_tensor_load_to_lds(g0, g1, z4, z4, 0);
#endif
            __builtin_amdgcn_s_wait_tensorcnt(0);
        }
        __syncthreads();
        for (int idx = tid; idx < 32 * CPAD; idx += 256)
            Xh[idx] = (_Float16)Xf[idx];
    } else {
        // ---- generic path with channel zero-padding ----
        for (int idx = tid; idx < 32 * CPAD; idx += 256) {
            int p = idx / CPAD, c = idx % CPAD;
            float v = (c < CIN) ? src[((size_t)g * 32 + p) * CIN + c] : 0.0f;
            Xf[idx] = v;
            Xh[idx] = (_Float16)v;
        }
    }
    __syncthreads();

    // knn-8 per point over feature space (tie -> smaller index)
    if (tid < 32) {
        const float* xp = Xf + tid * CPAD;
        for (int q = 0; q < 32; ++q) {
            const float* xq = Xf + q * CPAD;
            float s = 0.0f;
            for (int c = 0; c < CIN; ++c) { float d = xp[c] - xq[c]; s += d * d; }
            dd[tid * 32 + q] = s;
        }
        float* row = dd + tid * 32;
        for (int s = 0; s < KG_; ++s) {
            float bv = 3e38f; int bi = 0;
            for (int q = 0; q < 32; ++q) {
                float v = row[q];
                if (v < bv) { bv = v; bi = q; }
            }
            kn[tid * KG_ + s] = bi;
            row[bi] = 3e38f;
        }
    }
    __syncthreads();

    // WMMA GEMM: 2 row-tiles x (2D/16) col-tiles, K accumulated in 32-steps.
    const int COLT  = TWO_D / 16;
    const int TILES = 2 * COLT;
    const int wave  = tid >> 5;
    const int lane  = tid & 31;
    const int lh    = lane & 15;
    const int ksel  = (lane & 16) >> 1;  // +8 for upper half-wave
    for (int t = wave; t < TILES; t += 8) {
        int rt = t / COLT, ct = t % COLT;
        int r0 = rt * 16, c0 = ct * 16;
        v8f acc;
#pragma unroll
        for (int i = 0; i < 8; ++i) acc[i] = 0.0f;
#pragma unroll
        for (int ks = 0; ks < CPAD / 32; ++ks) {
            int k0 = ks * 32;
            v16h av, bmat;
#pragma unroll
            for (int h = 0; h < 16; ++h) {
                // A 16x32 f16 layout: K = h + (h&8) + (lane>=16 ? 8 : 0)
                int ka = k0 + h + (h & 8) + ksel;
                av[h] = Xh[(r0 + lh) * CPAD + ka];
                // B 32x16 f16 layout: K = h + (h&8) + (lane>=16 ? 8 : 0), N = lane&15
                int kb = k0 + h + (h & 8) + ksel;
                bmat[h] = Wp[kb * TWO_D + c0 + lh];
            }
            acc = __builtin_amdgcn_wmma_f32_16x16x32_f16(
                false, av, false, bmat, (short)0, acc, false, false);
        }
#pragma unroll
        for (int i = 0; i < 8; ++i)
            Y[(r0 + i + ksel) * TWO_D + c0 + lh] = acc[i];  // D f32 layout
    }
    __syncthreads();

    // relu + max over 8 neighbors
    for (int e = tid; e < 32 * DOUT; e += 256) {
        int p = e / DOUT, d = e % DOUT;
        float cb = Y[p * TWO_D + DOUT + d];
        float ga = gamma[d], be = beta[d];
        float mx = 0.0f;  // relu outputs are >= 0
#pragma unroll
        for (int s = 0; s < KG_; ++s) {
            int j = kn[p * KG_ + s];
            float hv = ga * (Y[j * TWO_D + d] + cb) + be;
            mx = fmaxf(mx, fmaxf(hv, 0.0f));
        }
        if (OUTMODE == 0) {
            dst[((size_t)g * 32 + p) * DOUT + d] = mx;
        } else {  // emb layout (b, d, m, p)
            int b = g >> 9, m = g & 511;
            dst[(((size_t)b * DOUT + d) * M_ + m) * K_ + p] = mx;
        }
    }
}

// ---------------------------------------------------------------------------
// 7) Attention: x1 = max_d emb, softmax over K, keypoints = sum aw * neigh_xyz
// ---------------------------------------------------------------------------
__global__ __launch_bounds__(32) void attn_kernel(const float* __restrict__ emb,
                                                  const float* __restrict__ clus,
                                                  float* __restrict__ aw,
                                                  float* __restrict__ kp) {
    __shared__ float sh[32];
    const int g = blockIdx.x;
    const int p = threadIdx.x;
    const int b = g >> 9, m = g & 511;
    const float* ep = emb + ((size_t)b * C3_ * M_ + m) * K_ + p;
    float mx = -3e38f;
    for (int d = 0; d < C3_; ++d) mx = fmaxf(mx, ep[(size_t)d * (M_ * K_)]);
    sh[p] = mx;
    __syncthreads();
    float wm = sh[0];
    for (int i = 1; i < 32; ++i) wm = fmaxf(wm, sh[i]);
    float e = expf(mx - wm);
    __syncthreads();
    sh[p] = e;
    __syncthreads();
    float sum = 0.0f;
    for (int i = 0; i < 32; ++i) sum += sh[i];
    float a = e / sum;
    aw[g * K_ + p] = a;
    for (int c = 0; c < 3; ++c) {
        __syncthreads();
        sh[p] = a * clus[((size_t)g * K_ + p) * 12 + 4 + c];
        __syncthreads();
        if (p == 0) {
            float t = 0.0f;
            for (int i = 0; i < 32; ++i) t += sh[i];
            kp[((size_t)b * 3 + c) * M_ + m] = t;
        }
    }
}

// ---------------------------------------------------------------------------
// 8) afm = emb * aw (in place in d_out), gcf = sum_k afm
// ---------------------------------------------------------------------------
__global__ void scale_gcf_kernel(float* __restrict__ afm,
                                 const float* __restrict__ aw,
                                 float* __restrict__ gcf) {
    int rid = blockIdx.x * blockDim.x + threadIdx.x;  // (b, c, m)
    if (rid >= B_ * C3_ * M_) return;
    int bm = (rid / (C3_ * M_)) * M_ + (rid % M_);
    const float* ap = aw + bm * K_;
    float* row = afm + (size_t)rid * K_;
    float sum = 0.0f;
#pragma unroll
    for (int k = 0; k < K_; ++k) {
        float v = row[k] * ap[k];
        row[k] = v;
        sum += v;
    }
    gcf[rid] = sum;
}

// ---------------------------------------------------------------------------
// 9) Per-node MLP 256->256->256->1, softplus + 0.001
// ---------------------------------------------------------------------------
__global__ __launch_bounds__(256) void mlp_kernel(const float* __restrict__ gcf,
                                                  const float* __restrict__ w1,
                                                  const float* __restrict__ bias1,
                                                  const float* __restrict__ gmm1,
                                                  const float* __restrict__ bb1,
                                                  const float* __restrict__ w2,
                                                  const float* __restrict__ bias2,
                                                  const float* __restrict__ gmm2,
                                                  const float* __restrict__ bb2,
                                                  const float* __restrict__ w3,
                                                  const float* __restrict__ bias3,
                                                  float* __restrict__ sal) {
    __shared__ float va[256], vb[256], red[256];
    const int g = blockIdx.x;
    const int d = threadIdx.x;
    const int b = g >> 9, m = g & 511;
    va[d] = gcf[((size_t)b * C3_ + d) * M_ + m];
    __syncthreads();
    float acc = 0.0f;
    for (int c = 0; c < C3_; ++c) acc += va[c] * w1[c * C3_ + d];
    vb[d] = fmaxf(gmm1[d] * (acc + bias1[d]) + bb1[d], 0.0f);
    __syncthreads();
    acc = 0.0f;
    for (int c = 0; c < C3_; ++c) acc += vb[c] * w2[c * C3_ + d];
    float h = fmaxf(gmm2[d] * (acc + bias2[d]) + bb2[d], 0.0f);
    red[d] = h * w3[d];
    __syncthreads();
    for (int s = 128; s > 0; s >>= 1) {
        if (d < s) red[d] += red[d + s];
        __syncthreads();
    }
    if (d == 0) {
        float sv = red[0] + bias3[0];
        float sp = fmaxf(sv, 0.0f) + log1pf(expf(-fabsf(sv)));
        sal[(size_t)b * M_ + m] = sp + 0.001f;
    }
}

// ---------------------------------------------------------------------------
extern "C" void kernel_launch(void* const* d_in, const int* in_sizes, int n_in,
                              void* d_out, int out_size, void* d_ws, size_t ws_size,
                              hipStream_t stream) {
    (void)in_sizes; (void)n_in; (void)out_size; (void)ws_size;
    const float* x    = (const float*)d_in[0];
    const int*   ridx = (const int*)d_in[1];
    const float* we1  = (const float*)d_in[2];
    const float* g1   = (const float*)d_in[3];
    const float* b1   = (const float*)d_in[4];
    const float* we2  = (const float*)d_in[5];
    const float* g2   = (const float*)d_in[6];
    const float* b2   = (const float*)d_in[7];
    const float* wm1  = (const float*)d_in[8];
    const float* bm1  = (const float*)d_in[9];
    const float* gm1  = (const float*)d_in[10];
    const float* bbm1 = (const float*)d_in[11];
    const float* wm2  = (const float*)d_in[12];
    const float* bm2  = (const float*)d_in[13];
    const float* gm2  = (const float*)d_in[14];
    const float* bbm2 = (const float*)d_in[15];
    const float* wm3  = (const float*)d_in[16];
    const float* bm3  = (const float*)d_in[17];

    float* out = (float*)d_out;
    float* kp  = out;              // (4,3,512)
    float* sal = out + 6144;       // (4,512)
    float* rc  = out + 8192;       // (4,12,512,32)
    float* afm = out + 794624;     // (4,256,512,32)

    char* ws = (char*)d_ws;
    size_t off = 0;
    auto carve = [&](size_t bytes) {
        char* p = ws + off;
        off = (off + bytes + 255) & ~(size_t)255;
        return p;
    };
    int*      node_idx = (int*)carve((size_t)B_ * M_ * 4);
    float*    node     = (float*)carve((size_t)B_ * M_ * 4 * 4);
    int*      sel      = (int*)carve((size_t)B_ * M_ * K_ * 4);
    float*    clus     = (float*)carve((size_t)B_ * M_ * K_ * 12 * 4);
    float*    h1       = (float*)carve((size_t)B_ * M_ * K_ * 64 * 4);
    float*    awb      = (float*)carve((size_t)B_ * M_ * K_ * 4);
    float*    gcf      = (float*)carve((size_t)B_ * C3_ * M_ * 4);
    _Float16* w1p      = (_Float16*)carve((size_t)32 * 128 * 2);
    _Float16* w2p      = (_Float16*)carve((size_t)64 * 512 * 2);

    prep_weights_kernel<<<144, 256, 0, stream>>>(we1, we2, w1p, w2p);
    fps_kernel<<<B_, 1024, 0, stream>>>((const float4*)x, node_idx);
    gather_node_kernel<<<8, 256, 0, stream>>>((const float4*)x, node_idx, (float4*)node);

    size_t shm_knn = (size_t)N_ * 4;
    knn64_kernel<<<B_ * M_, 256, shm_knn, stream>>>((const float4*)x, (const float4*)node,
                                                    ridx, sel);
    cluster_kernel<<<(B_ * M_ * K_) / 256, 256, 0, stream>>>((const float4*)x,
                                                             (const float4*)node, sel, clus, rc);

    size_t shm1 = (size_t)32 * 32 * 4 + 32 * 32 * 2 + 32 * 128 * 4 + 32 * 32 * 4 + 32 * 8 * 4;
    edgeconv_kernel<12, 32, 64, 0><<<B_ * M_, 256, shm1, stream>>>(clus, w1p, g1, b1, h1);

    size_t shm2 = (size_t)32 * 64 * 4 + 32 * 64 * 2 + 32 * 512 * 4 + 32 * 32 * 4 + 32 * 8 * 4;
    edgeconv_kernel<64, 64, 256, 1><<<B_ * M_, 256, shm2, stream>>>(h1, w2p, g2, b2, afm);

    attn_kernel<<<B_ * M_, 32, 0, stream>>>(afm, clus, awb, kp);
    scale_gcf_kernel<<<(B_ * C3_ * M_ + 255) / 256, 256, 0, stream>>>(afm, awb, gcf);
    mlp_kernel<<<B_ * M_, 256, 0, stream>>>(gcf, wm1, bm1, gm1, bbm1,
                                            wm2, bm2, gm2, bbm2, wm3, bm3, sal);
}
